// ClusterGCN_21028159881632
// MI455X (gfx1250) — matrix operand order repaired and checked by reading, not compiled, not run on hardware
//
#include <hip/hip_runtime.h>
#include <hip/hip_bf16.h>

#define D 128
#define NLAYERS 3
#define EPS_BN 1e-5f
#define EPS_NORM 1e-12f

typedef __attribute__((ext_vector_type(16))) _Float16 v16h;
typedef __attribute__((ext_vector_type(8)))  float    v8f;

enum { EP_NONE = 0, EP_TANH = 1, EP_RELU = 2 };

// ---------------- degree / normalization coefficients ----------------

__global__ void k_deg(const int* __restrict__ dst, float* __restrict__ deg, int E) {
    int i = blockIdx.x * blockDim.x + threadIdx.x;
    if (i < E) atomicAdd(&deg[dst[i]], 1.0f);
}

__global__ void k_dinv(float* __restrict__ deg, int n) {
    int i = blockIdx.x * blockDim.x + threadIdx.x;
    if (i < n) deg[i] = rsqrtf(deg[i] + 1.0f);   // +1 self loop
}

// ---------------- WMMA GEMM: C[N,128] = A_f32[N,128] @ B_f32[128,128] (+bias, act) ----------------
// fp32 inputs are converted to f16 fragments in registers (fp32 accumulate via WMMA).
// Block = 256 threads = 8 waves; wave ct owns cols [ct*16, ct*16+16).
// Block processes 4 row-tiles (64 rows); B fragments (weights) stay register-resident.
// Fragment layouts per CDNA5 ISA 7.12.2 (wave32):
//   A 16x32 f16 : lanes 0-15 -> M=lane, K={0..7,16..23}; lanes 16-31 -> M=lane-16, K={8..15,24..31}
//   B 32x16 f16 : lane L -> N=L%16, K as for A (16 K-values across 8 VGPRs)
//   D 16x16 f32 : VGPR j -> M = j + 8*(lane/16), N = lane%16
template <int MODE, bool HAS_BIAS>
__global__ __launch_bounds__(256) void k_gemm_wmma(
    const float* __restrict__ A, const float* __restrict__ B,
    const float* __restrict__ bias, float* __restrict__ C, int nrows) {
    const int lane = threadIdx.x & 31;
    const int ct   = threadIdx.x >> 5;       // column tile 0..7
    const int half = lane >> 4;
    const int m    = lane & 15;
    const int ncol = lane & 15;
    const int blockRow0 = blockIdx.x * 64;

    // Preload & convert all 4 K-step B fragments once (weights, reused across row tiles).
    v16h bfrag[4];
    #pragma unroll
    for (int t = 0; t < 4; ++t) {
        const float* bp = B + (size_t)(t * 32) * D + ct * 16 + ncol;
        #pragma unroll
        for (int j = 0; j < 8; ++j) {
            bfrag[t][j]     = (_Float16)bp[(size_t)(half * 8 + j) * D];
            bfrag[t][j + 8] = (_Float16)bp[(size_t)(16 + half * 8 + j) * D];
        }
    }
    const float bcol = HAS_BIAS ? bias[ct * 16 + ncol] : 0.0f;

    #pragma unroll
    for (int rt = 0; rt < 4; ++rt) {
        const int rowBase = blockRow0 + rt * 16;
        int arow = rowBase + m;
        if (arow >= nrows) arow = nrows - 1;   // branchless clamp; stores guarded below

        v8f acc = {};
        #pragma unroll
        for (int t = 0; t < 4; ++t) {
            const float* ap = A + (size_t)arow * D + t * 32;
            v16h a;
            #pragma unroll
            for (int j = 0; j < 8; ++j) {
                a[j]     = (_Float16)ap[half * 8 + j];
                a[j + 8] = (_Float16)ap[16 + half * 8 + j];
            }
            acc = __builtin_amdgcn_wmma_f32_16x16x32_f16(
                /*neg_a=*/false, a, /*neg_b=*/false, bfrag[t],
                /*c_mod=*/(short)0, acc, /*reuse_a=*/false, /*reuse_b=*/false);
        }

        #pragma unroll
        for (int j = 0; j < 8; ++j) {
            const int row = rowBase + j + 8 * half;
            if (row < nrows) {
                float v = acc[j] + bcol;
                if (MODE == EP_TANH)      v = tanhf(v);
                else if (MODE == EP_RELU) v = fmaxf(v, 0.0f);
                C[(size_t)row * D + ct * 16 + ncol] = v;
            }
        }
    }
}

static inline void launch_gemm(const float* A, const float* B, const float* bias,
                               float* C, int nrows, int mode, hipStream_t s) {
    const int grid = (nrows + 63) / 64;
    if (bias == nullptr) {
        k_gemm_wmma<EP_NONE, false><<<grid, 256, 0, s>>>(A, B, nullptr, C, nrows);
    } else if (mode == EP_TANH) {
        k_gemm_wmma<EP_TANH, true><<<grid, 256, 0, s>>>(A, B, bias, C, nrows);
    } else if (mode == EP_RELU) {
        k_gemm_wmma<EP_RELU, true><<<grid, 256, 0, s>>>(A, B, bias, C, nrows);
    } else {
        k_gemm_wmma<EP_NONE, true><<<grid, 256, 0, s>>>(A, B, bias, C, nrows);
    }
}

// ---------------- edge scatter: agg[dst] += lin[src] * dinv[src]*dinv[dst] ----------------
__global__ __launch_bounds__(128) void k_scatter(
    const float* __restrict__ lin, const int* __restrict__ src, const int* __restrict__ dst,
    const float* __restrict__ dinv, float* __restrict__ agg, int E) {
    const int c = threadIdx.x;               // channel 0..127
    for (int e = blockIdx.x; e < E; e += gridDim.x) {
        const int s = src[e];
        const int d = dst[e];
        const float coef = dinv[s] * dinv[d];
        atomicAdd(&agg[(size_t)d * D + c], lin[(size_t)s * D + c] * coef);
    }
}

// ---------------- combine: h = agg0 + lin0*dinv0^2 + b0 + agg1 + lin1*dinv1^2 + b1 ----------------
__global__ void k_combine(const float* __restrict__ agg0, const float* __restrict__ lin0,
                          const float* __restrict__ dinv0, const float* __restrict__ b0,
                          const float* __restrict__ agg1, const float* __restrict__ lin1,
                          const float* __restrict__ dinv1, const float* __restrict__ b1,
                          float* __restrict__ out, int total4) {
    int i4 = blockIdx.x * blockDim.x + threadIdx.x;   // float4 index
    if (i4 >= total4) return;
    const int base = i4 * 4;
    const int n = base >> 7;
    const int c = base & (D - 1);
    const float i0 = dinv0[n], s0 = i0 * i0;
    const float i1 = dinv1[n], s1 = i1 * i1;
    const float4 a0 = *(const float4*)(agg0 + base);
    const float4 l0 = *(const float4*)(lin0 + base);
    const float4 a1 = *(const float4*)(agg1 + base);
    const float4 l1 = *(const float4*)(lin1 + base);
    const float4 v0 = *(const float4*)(b0 + c);
    const float4 v1 = *(const float4*)(b1 + c);
    float4 o;
    o.x = a0.x + l0.x * s0 + v0.x + a1.x + l1.x * s1 + v1.x;
    o.y = a0.y + l0.y * s0 + v0.y + a1.y + l1.y * s1 + v1.y;
    o.z = a0.z + l0.z * s0 + v0.z + a1.z + l1.z * s1 + v1.z;
    o.w = a0.w + l0.w * s0 + v0.w + a1.w + l1.w * s1 + v1.w;
    *(float4*)(out + base) = o;
}

// ---------------- batchnorm (training-mode batch stats, biased var) ----------------
__global__ __launch_bounds__(128) void k_bn_stats(const float* __restrict__ x, int nrows,
                                                  float* __restrict__ sum, float* __restrict__ sumsq) {
    const int c = threadIdx.x;
    float s = 0.0f, s2 = 0.0f;
    for (int r = blockIdx.x; r < nrows; r += gridDim.x) {
        const float v = x[(size_t)r * D + c];
        s += v; s2 += v * v;
    }
    atomicAdd(&sum[c], s);
    atomicAdd(&sumsq[c], s2);
}

template <bool RELU>
__global__ void k_bn_apply(float* __restrict__ x, int total4, int nrows,
                           const float* __restrict__ sum, const float* __restrict__ sumsq,
                           const float* __restrict__ g, const float* __restrict__ bt) {
    int i4 = blockIdx.x * blockDim.x + threadIdx.x;
    if (i4 >= total4) return;
    const int base = i4 * 4;
    const int c = base & (D - 1);
    const float invN = 1.0f / (float)nrows;
    const float4 sm  = *(const float4*)(sum + c);
    const float4 sq  = *(const float4*)(sumsq + c);
    const float4 gg  = *(const float4*)(g + c);
    const float4 bb  = *(const float4*)(bt + c);
    float4 v = *(const float4*)(x + base);
    float mu, var;
    mu = sm.x * invN; var = sq.x * invN - mu * mu; v.x = (v.x - mu) * rsqrtf(var + EPS_BN) * gg.x + bb.x;
    mu = sm.y * invN; var = sq.y * invN - mu * mu; v.y = (v.y - mu) * rsqrtf(var + EPS_BN) * gg.y + bb.y;
    mu = sm.z * invN; var = sq.z * invN - mu * mu; v.z = (v.z - mu) * rsqrtf(var + EPS_BN) * gg.z + bb.z;
    mu = sm.w * invN; var = sq.w * invN - mu * mu; v.w = (v.w - mu) * rsqrtf(var + EPS_BN) * gg.w + bb.w;
    if (RELU) {
        v.x = fmaxf(v.x, 0.0f); v.y = fmaxf(v.y, 0.0f);
        v.z = fmaxf(v.z, 0.0f); v.w = fmaxf(v.w, 0.0f);
    }
    *(float4*)(x + base) = v;
}

// ---------------- row-wise L2 normalization (one wave32 per row) ----------------
__global__ __launch_bounds__(256) void k_l2norm(const float* __restrict__ in,
                                                float* __restrict__ out, int nrows) {
    const int wave = threadIdx.x >> 5;
    const int lane = threadIdx.x & 31;
    const int row  = blockIdx.x * 8 + wave;
    if (row >= nrows) return;
    const float4 v = *(const float4*)(in + (size_t)row * D + lane * 4);
    float ss = v.x * v.x + v.y * v.y + v.z * v.z + v.w * v.w;
    #pragma unroll
    for (int off = 16; off >= 1; off >>= 1) ss += __shfl_xor(ss, off);
    const float inv = 1.0f / fmaxf(sqrtf(ss), EPS_NORM);
    float4 o; o.x = v.x * inv; o.y = v.y * inv; o.z = v.z * inv; o.w = v.w * inv;
    *(float4*)(out + (size_t)row * D + lane * 4) = o;
}

// ---------------- orchestration ----------------

extern "C" void kernel_launch(void* const* d_in, const int* in_sizes, int n_in,
                              void* d_out, int out_size, void* d_ws, size_t ws_size,
                              hipStream_t stream) {
    const float* x      = (const float*)d_in[0];
    const int*   ei0    = (const int*)  d_in[1];
    const int*   ei1    = (const int*)  d_in[2];
    const float* W0     = (const float*)d_in[3];
    const float* b0     = (const float*)d_in[4];
    const float* W1     = (const float*)d_in[5];
    const float* b1     = (const float*)d_in[6];
    const float* gamma  = (const float*)d_in[7];
    const float* beta   = (const float*)d_in[8];
    const float* emb1_W = (const float*)d_in[9];
    const float* emb1_b = (const float*)d_in[10];
    const float* emb2_W = (const float*)d_in[11];
    const float* emb2_b = (const float*)d_in[12];
    const float* ph1_Wa = (const float*)d_in[13];
    const float* ph1_ba = (const float*)d_in[14];
    const float* ph1_Wb = (const float*)d_in[15];
    const float* ph1_bb = (const float*)d_in[16];
    const float* ph2_Wa = (const float*)d_in[17];
    const float* ph2_ba = (const float*)d_in[18];
    const float* ph2_Wb = (const float*)d_in[19];
    const float* ph2_bb = (const float*)d_in[20];

    const int N  = in_sizes[0] / D;
    const int E0 = in_sizes[1] / 2;
    const int E1 = in_sizes[2] / 2;
    const size_t ND = (size_t)N * D;

    // workspace carve-out (256B aligned slabs)
    char* wp = (char*)d_ws;
    auto carve = [&](size_t bytes) { char* p = wp; wp += (bytes + 255) & ~(size_t)255; return p; };
    float* hcur  = (float*)carve(ND * sizeof(float));
    float* lin0  = (float*)carve(ND * sizeof(float));
    float* lin1  = (float*)carve(ND * sizeof(float));
    float* agg0  = (float*)carve(ND * sizeof(float));
    float* agg1  = (float*)carve(ND * sizeof(float));
    float* tmp   = (float*)carve(ND * sizeof(float));
    float* dinv0 = (float*)carve((size_t)N * sizeof(float));
    float* dinv1 = (float*)carve((size_t)N * sizeof(float));
    float* bnsum = (float*)carve(D * sizeof(float));
    float* bnsq  = (float*)carve(D * sizeof(float));

    float* out = (float*)d_out;
    float* e1  = out;              // [N,D]
    float* e2  = out + ND;         // [N,D]
    float* p1  = out + 2 * ND;     // [N,D]
    float* p2  = out + 3 * ND;     // [N,D]

    const int TB = 256;
    const int total4 = (int)(ND / 4);
    const int gV4  = (total4 + TB - 1) / TB;
    const int gL2  = (N + 7) / 8;

    const int* src0 = ei0;      const int* dst0 = ei0 + E0;
    const int* src1 = ei1;      const int* dst1 = ei1 + E1;

    // ---- degrees & dinv (invariant across layers) ----
    hipMemsetAsync(dinv0, 0, (size_t)N * sizeof(float), stream);
    hipMemsetAsync(dinv1, 0, (size_t)N * sizeof(float), stream);
    k_deg<<<(E0 + TB - 1) / TB, TB, 0, stream>>>(dst0, dinv0, E0);
    k_deg<<<(E1 + TB - 1) / TB, TB, 0, stream>>>(dst1, dinv1, E1);
    k_dinv<<<(N + TB - 1) / TB, TB, 0, stream>>>(dinv0, N);
    k_dinv<<<(N + TB - 1) / TB, TB, 0, stream>>>(dinv1, N);

    // ---- h = x ----
    hipMemcpyAsync(hcur, x, ND * sizeof(float), hipMemcpyDeviceToDevice, stream);

    // ---- GCN layers ----
    for (int i = 0; i < NLAYERS; ++i) {
        launch_gemm(hcur, W0 + (size_t)i * D * D, nullptr, lin0, N, EP_NONE, stream);
        launch_gemm(hcur, W1 + (size_t)i * D * D, nullptr, lin1, N, EP_NONE, stream);

        hipMemsetAsync(agg0, 0, ND * sizeof(float), stream);
        hipMemsetAsync(agg1, 0, ND * sizeof(float), stream);
        k_scatter<<<E0, D, 0, stream>>>(lin0, src0, dst0, dinv0, agg0, E0);
        k_scatter<<<E1, D, 0, stream>>>(lin1, src1, dst1, dinv1, agg1, E1);

        k_combine<<<gV4, TB, 0, stream>>>(agg0, lin0, dinv0, b0 + (size_t)i * D,
                                          agg1, lin1, dinv1, b1 + (size_t)i * D,
                                          hcur, total4);

        hipMemsetAsync(bnsum, 0, D * sizeof(float), stream);
        hipMemsetAsync(bnsq,  0, D * sizeof(float), stream);
        k_bn_stats<<<256, D, 0, stream>>>(hcur, N, bnsum, bnsq);
        if (i < NLAYERS - 1)
            k_bn_apply<true><<<gV4, TB, 0, stream>>>(hcur, total4, N, bnsum, bnsq,
                                                     gamma + (size_t)i * D, beta + (size_t)i * D);
        else
            k_bn_apply<false><<<gV4, TB, 0, stream>>>(hcur, total4, N, bnsum, bnsq,
                                                      gamma + (size_t)i * D, beta + (size_t)i * D);
    }

    // ---- heads ----
    // e1 = tanh(h @ emb1_W + b)
    launch_gemm(hcur, emb1_W, emb1_b, e1, N, EP_TANH, stream);

    // e2 = l2norm(tanh(h @ emb2_W + b))
    launch_gemm(hcur, emb2_W, emb2_b, tmp, N, EP_TANH, stream);
    k_l2norm<<<gL2, TB, 0, stream>>>(tmp, e2, N);

    // p1 = l2norm(relu(e1 @ Wa + ba) @ Wb + bb)
    launch_gemm(e1, ph1_Wa, ph1_ba, tmp, N, EP_RELU, stream);
    launch_gemm(tmp, ph1_Wb, ph1_bb, agg0, N, EP_NONE, stream);   // reuse agg0 as scratch
    k_l2norm<<<gL2, TB, 0, stream>>>(agg0, p1, N);

    // p2 = l2norm(relu(e2 @ Wa + ba) @ Wb + bb)
    launch_gemm(e2, ph2_Wa, ph2_ba, tmp, N, EP_RELU, stream);
    launch_gemm(tmp, ph2_Wb, ph2_bb, agg0, N, EP_NONE, stream);
    k_l2norm<<<gL2, TB, 0, stream>>>(agg0, p2, N);

    (void)n_in; (void)out_size; (void)ws_size;
}
